// MultiAgentGRULoop_1683627180447
// MI455X (gfx1250) — matrix operand order repaired and checked by reading, not compiled, not run on hardware
//
#include <hip/hip_runtime.h>
#include <math.h>

// ---------------------------------------------------------------------------
// Multi-agent 3-layer GRU on MI455X (gfx1250), bf16 WMMA with fp32 accumulate.
// Layer-by-layer processing: per layer, 100 sequential fused step kernels
// (both GEMMs + gates). Activations/weights converted once to bf16; recurrent
// state kept in fp32 (epilogue path) + bf16 mirror (WMMA A operand),
// double-buffered across time steps to avoid cross-wave races.
// ---------------------------------------------------------------------------

typedef __attribute__((ext_vector_type(16))) __bf16 v16bf;
typedef __attribute__((ext_vector_type(8)))  __bf16 v8bf;
typedef __attribute__((ext_vector_type(8)))  float  v8f;

static constexpr int Bc  = 8;
static constexpr int Tc  = 100;
static constexpr int Ac  = 64;
static constexpr int Hc  = 512;
static constexpr int Lc  = 3;
static constexpr int BAc = Bc * Ac;   // 512

__device__ __forceinline__ float sigmoid_f(float x) {
  return 1.0f / (1.0f + __expf(-x));
}

// A-matrix 16x32 bf16 k order used here (B uses the same): lane holds row M=lane%16;
// elements 0..15 = K starting at (lane<16 ? 0 : 16), contiguous.
//   -> two contiguous 16B loads.
__device__ __forceinline__ v16bf load_a_tile(const __bf16* __restrict__ p) {
  v8bf lo = *(const v8bf*)(p);
  v8bf hi = *(const v8bf*)(p + 8);
  return __builtin_shufflevector(lo, hi, 0, 1, 2, 3, 4, 5, 6, 7,
                                         8, 9, 10, 11, 12, 13, 14, 15);
}

__device__ __forceinline__ v8f wmma_bf16(v16bf a, v16bf b, v8f c) {
  return __builtin_amdgcn_wmma_f32_16x16x32_bf16(
      /*neg_a=*/false, a, /*neg_b=*/false, b,
      /*c_mod=*/(short)0, c, /*reuse_a=*/false, /*reuse_b=*/false);
}

// ---------------------------------------------------------------------------
// Prep kernels
// ---------------------------------------------------------------------------
__global__ void cvt_w_to_bf16(const float* __restrict__ w,
                              __bf16* __restrict__ wb, int n) {
  int i = blockIdx.x * blockDim.x + threadIdx.x;
  if (i < n) wb[i] = (__bf16)w[i];
}

// x: [B,T,A,H] fp32  ->  act: [T, B*A, H] bf16
__global__ void reorder_x_to_bf16(const float* __restrict__ x,
                                  __bf16* __restrict__ act) {
  int idx = blockIdx.x * blockDim.x + threadIdx.x;
  if (idx >= Tc * BAc * Hc) return;
  int h  = idx % Hc;
  int r  = idx / Hc;
  int ba = r % BAc;
  int t  = r / BAc;
  int b  = ba / Ac;
  int a  = ba % Ac;
  act[idx] = (__bf16)x[(((size_t)b * Tc + t) * Ac + a) * Hc + h];
}

// ---------------------------------------------------------------------------
// Fused GRU step: one 16x16 output tile per wave.
//   gi = actIn @ wih^T ; gh = hPrev @ whh^T  (bf16 WMMA, f32 accum)
//   r = sig(gi_r+gh_r+b), z = sig(gi_z+gh_z+b), n = tanh(gi_n + b + r*(gh_n+b))
//   h = (1-z)*n + z*h_old ; masked by valid[b,t,a]
// ---------------------------------------------------------------------------
__global__ __launch_bounds__(128)
void gru_step_wmma(const __bf16* __restrict__ actIn,   // [BA,H] at time t (bf16)
                   const __bf16* __restrict__ hbfIn,   // [BA,H] h(t-1) bf16
                   const __bf16* __restrict__ wih,     // [3H,H] bf16
                   const __bf16* __restrict__ whh,     // [3H,H] bf16
                   const float*  __restrict__ bih,     // [3H]
                   const float*  __restrict__ bhh,     // [3H]
                   const int*    __restrict__ valid,   // [B,T,A]
                   float*        __restrict__ hf32,    // [BA,H] fp32 state (in/out, lane-private cells)
                   __bf16*       __restrict__ hbfOut,  // [BA,H] h(t) bf16
                   __bf16*       __restrict__ actOut,  // [BA,H] next-layer input (or null)
                   float*        __restrict__ yOut,    // [B,T,A,H] final output (or null)
                   int t) {
  const int lane = threadIdx.x & 31;
  const int wave = threadIdx.x >> 5;
  const int tile = blockIdx.x * 4 + wave;   // 0..1023
  const int tm   = tile >> 5;               // BA row tile, 0..31
  const int tn   = tile & 31;               // H  col tile, 0..31
  const int lh   = lane & 15;
  const int sel  = lane >> 4;               // 0 or 1 (lane half)

  const int rowA = tm * 16 + lh;            // A-operand row for this lane
  const int colN = tn * 16 + lh;            // output column n (within H)

  // A operands: activations and previous hidden state (both bf16, row-major [BA,H])
  const __bf16* pAx = actIn + (size_t)rowA * Hc + sel * 16;
  const __bf16* pAh = hbfIn + (size_t)rowA * Hc + sel * 16;
  // B operands: weight rows (gi[:,j] = sum_k in[:,k]*W[j,k]) — contiguous in K.
  const __bf16* pIr = wih + ((size_t)(0 * Hc + colN)) * Hc + sel * 16;
  const __bf16* pIz = wih + ((size_t)(1 * Hc + colN)) * Hc + sel * 16;
  const __bf16* pIn = wih + ((size_t)(2 * Hc + colN)) * Hc + sel * 16;
  const __bf16* pHr = whh + ((size_t)(0 * Hc + colN)) * Hc + sel * 16;
  const __bf16* pHz = whh + ((size_t)(1 * Hc + colN)) * Hc + sel * 16;
  const __bf16* pHn = whh + ((size_t)(2 * Hc + colN)) * Hc + sel * 16;

  v8f c_ir = {}, c_iz = {}, c_in = {}, c_hr = {}, c_hz = {}, c_hn = {};

#pragma unroll 4
  for (int k = 0; k < Hc; k += 32) {
    v16bf ax = load_a_tile(pAx + k);
    v16bf ah = load_a_tile(pAh + k);
    v16bf bir = *(const v16bf*)(pIr + k);
    v16bf biz = *(const v16bf*)(pIz + k);
    v16bf bin = *(const v16bf*)(pIn + k);
    v16bf bhr = *(const v16bf*)(pHr + k);
    v16bf bhz = *(const v16bf*)(pHz + k);
    v16bf bhn = *(const v16bf*)(pHn + k);
    c_ir = wmma_bf16(ax, bir, c_ir);
    c_iz = wmma_bf16(ax, biz, c_iz);
    c_in = wmma_bf16(ax, bin, c_in);
    c_hr = wmma_bf16(ah, bhr, c_hr);
    c_hz = wmma_bf16(ah, bhz, c_hz);
    c_hn = wmma_bf16(ah, bhn, c_hn);
  }

  // Biases for this output column
  const float bir_s = bih[colN],          bhr_s = bhh[colN];
  const float biz_s = bih[Hc + colN],     bhz_s = bhh[Hc + colN];
  const float bin_s = bih[2 * Hc + colN], bhn_s = bhh[2 * Hc + colN];

  // C/D layout: VGPR v -> row tm*16 + sel*8 + v, col = colN
#pragma unroll
  for (int v = 0; v < 8; ++v) {
    const int m  = tm * 16 + sel * 8 + v;      // BA row index
    const int b  = m >> 6;                     // / Ac
    const int a  = m & 63;                     // % Ac
    const size_t hidx = (size_t)m * Hc + colN;

    float h_old = hf32[hidx];
    float r  = sigmoid_f(c_ir[v] + bir_s + c_hr[v] + bhr_s);
    float z  = sigmoid_f(c_iz[v] + biz_s + c_hz[v] + bhz_s);
    float ng = tanhf(c_in[v] + bin_s + r * (c_hn[v] + bhn_s));
    float hn = (1.0f - z) * ng + z * h_old;

    const int ok = valid[((size_t)b * Tc + t) * Ac + a];
    hn = ok ? hn : 0.0f;

    hf32[hidx]   = hn;
    hbfOut[hidx] = (__bf16)hn;
    if (actOut) actOut[hidx] = (__bf16)hn;
    if (yOut)   yOut[(((size_t)b * Tc + t) * Ac + a) * Hc + colN] = hn;
  }
}

// ---------------------------------------------------------------------------
// Host-side orchestration (all on `stream`, graph-capture safe)
// ---------------------------------------------------------------------------
extern "C" void kernel_launch(void* const* d_in, const int* in_sizes, int n_in,
                              void* d_out, int out_size, void* d_ws, size_t ws_size,
                              hipStream_t stream) {
  (void)in_sizes; (void)n_in; (void)out_size; (void)ws_size;

  const float* x     = (const float*)d_in[0];
  const int*   valid = (const int*)d_in[1];
  const float* w_ih  = (const float*)d_in[2];
  const float* w_hh  = (const float*)d_in[3];
  const float* b_ih  = (const float*)d_in[4];
  const float* b_hh  = (const float*)d_in[5];
  float* out = (float*)d_out;

  auto alignup = [](size_t v) { return (v + 255) & ~(size_t)255; };
  char* p = (char*)d_ws;
  const size_t wElems   = (size_t)Lc * 3 * Hc * Hc;   // per weight tensor
  const size_t actElems = (size_t)Tc * BAc * Hc;
  const size_t hElems   = (size_t)BAc * Hc;

  __bf16* wb_ih = (__bf16*)p; p += alignup(wElems * 2);
  __bf16* wb_hh = (__bf16*)p; p += alignup(wElems * 2);
  __bf16* act0  = (__bf16*)p; p += alignup(actElems * 2);
  __bf16* act1  = (__bf16*)p; p += alignup(actElems * 2);
  float*  hf32  = (float*)p;  p += alignup(hElems * 4);
  __bf16* hbf0  = (__bf16*)p; p += alignup(hElems * 2);
  __bf16* hbf1  = (__bf16*)p; p += alignup(hElems * 2);

  // One-time conversions (deterministic, re-run every call)
  const int wn = (int)wElems;
  cvt_w_to_bf16<<<(wn + 255) / 256, 256, 0, stream>>>(w_ih, wb_ih, wn);
  cvt_w_to_bf16<<<(wn + 255) / 256, 256, 0, stream>>>(w_hh, wb_hh, wn);
  const int xn = (int)actElems;
  reorder_x_to_bf16<<<(xn + 255) / 256, 256, 0, stream>>>(x, act0);

  __bf16* aIn  = act0;   // layer input  [T,BA,H] bf16
  __bf16* aOut = act1;   // layer output [T,BA,H] bf16

  for (int l = 0; l < Lc; ++l) {
    hipMemsetAsync(hf32, 0, hElems * 4, stream);
    hipMemsetAsync(hbf0, 0, hElems * 2, stream);
    __bf16* hin  = hbf0;
    __bf16* hout = hbf1;
    const bool last = (l == Lc - 1);
    const __bf16* wih_l = wb_ih + (size_t)l * 3 * Hc * Hc;
    const __bf16* whh_l = wb_hh + (size_t)l * 3 * Hc * Hc;
    const float*  bih_l = b_ih + (size_t)l * 3 * Hc;
    const float*  bhh_l = b_hh + (size_t)l * 3 * Hc;

    for (int t = 0; t < Tc; ++t) {
      gru_step_wmma<<<256, 128, 0, stream>>>(
          aIn + (size_t)t * BAc * Hc, hin, wih_l, whh_l, bih_l, bhh_l,
          valid, hf32, hout,
          last ? (__bf16*)nullptr : (aOut + (size_t)t * BAc * Hc),
          last ? out : (float*)nullptr, t);
      __bf16* tmp = hin; hin = hout; hout = tmp;
    }
    __bf16* ta = aIn; aIn = aOut; aOut = ta;
  }
}
